// CRF_48979807044150
// MI455X (gfx1250) — matrix operand order repaired and checked by reading, not compiled, not run on hardware
//
#include <hip/hip_runtime.h>
#include <cstdint>
#include <cstddef>

typedef __attribute__((ext_vector_type(16))) _Float16 v16h;
typedef __attribute__((ext_vector_type(8)))  float    v8f;

#define B_  512
#define T_  1024
#define K_  48
#define KP_ 64   // K padded to 2 WMMA k-chunks of 32

__device__ __forceinline__ float fast_exp(float x) {  // e^x via v_exp_f32 (exp2)
    return __builtin_amdgcn_exp2f(x * 1.4426950408889634f);
}
__device__ __forceinline__ float fast_log(float x) {  // ln(x) via v_log_f32 (log2)
    return __builtin_amdgcn_logf(x) * 0.6931471805599453f;
}

union V16HU { v16h v; unsigned u[8]; unsigned short s[16]; };

__device__ __forceinline__ unsigned pk2(float lo, float hi) {
    _Float16 l = (_Float16)lo, h = (_Float16)hi;
    unsigned short ul = __builtin_bit_cast(unsigned short, l);
    unsigned short uh = __builtin_bit_cast(unsigned short, h);
    return (unsigned)ul | ((unsigned)uh << 16);
}

// ---------------- Kernel 1: precompute maxT and E=exp(trans-maxT) as padded f16 --------
__global__ void crf_prep(const float* __restrict__ trans,
                         float* __restrict__ maxT_out,
                         unsigned short* __restrict__ Ef16) {
    __shared__ float red[256];
    const int tid = threadIdx.x;
    float mx = -3.0e38f;
    for (int i = tid; i < K_ * K_; i += 256) mx = fmaxf(mx, trans[i]);
    red[tid] = mx; __syncthreads();
    for (int s = 128; s > 0; s >>= 1) {
        if (tid < s) red[tid] = fmaxf(red[tid], red[tid + s]);
        __syncthreads();
    }
    const float M = red[0];
    if (tid == 0) maxT_out[0] = M;
    for (int idx = tid; idx < K_ * KP_; idx += 256) {
        const int i = idx >> 6, j = idx & (KP_ - 1);
        float v = (j < K_) ? fast_exp(trans[i * K_ + j] - M) : 0.0f;
        _Float16 h = (_Float16)v;
        Ef16[idx] = __builtin_bit_cast(unsigned short, h);
    }
}

// ---------------- Kernel 2: WMMA forward recursion, 1 wave per 16-batch tile -----------
__global__ __launch_bounds__(32, 1) void crf_forward(
        const float* __restrict__ emis, const int* __restrict__ lengths,
        const unsigned short* __restrict__ Ef16, const float* __restrict__ maxTp,
        const float* __restrict__ prior, const float* __restrict__ finalT,
        float* __restrict__ logZ) {
    const int lane = threadIdx.x;
    const int n    = lane & 15;   // batch-in-tile == N column of C/D
    const int hf   = lane >> 4;   // half-wave id
    const int b    = blockIdx.x * 16 + n;
    const float maxT = maxTp[0];
    int lenb = lengths[b]; if (lenb < 1) lenb = 1;
    const int t_last = lenb - 1;

    // Build constant A fragments (E matrix) once: A[m][c] covers rows m*16..+15, K=c*32..+31
    v16h A[3][2];
    #pragma unroll
    for (int m = 0; m < 3; ++m) {
        #pragma unroll
        for (int c = 0; c < 2; ++c) {
            V16HU av;
            #pragma unroll
            for (int h = 0; h < 16; ++h) {
                const int r = h >> 1, hi = h & 1;
                const int Kk = hf * 16 + 2 * r + hi;
                av.s[h] = Ef16[(m * 16 + n) * KP_ + c * 32 + Kk];
            }
            A[m][c] = av.v;
        }
    }

    const float* eb = emis + (size_t)b * T_ * K_;

    // t = 0: alpha = emis[:,0,:] + prior   (C/D layout: reg r -> row m*16 + hf*8 + r%.. )
    float a[3][8], af[3][8];
    #pragma unroll
    for (int m = 0; m < 3; ++m) {
        const float4 e0 = *(const float4*)(eb + m * 16 + hf * 8);
        const float4 e1 = *(const float4*)(eb + m * 16 + hf * 8 + 4);
        const float4 p0 = *(const float4*)(prior + m * 16 + hf * 8);
        const float4 p1 = *(const float4*)(prior + m * 16 + hf * 8 + 4);
        a[m][0] = e0.x + p0.x; a[m][1] = e0.y + p0.y; a[m][2] = e0.z + p0.z; a[m][3] = e0.w + p0.w;
        a[m][4] = e1.x + p1.x; a[m][5] = e1.y + p1.y; a[m][6] = e1.z + p1.z; a[m][7] = e1.w + p1.w;
        #pragma unroll
        for (int r = 0; r < 8; ++r) af[m][r] = a[m][r];   // snapshot covers t_last==0
    }

    for (int t = 1; t < T_; ++t) {
        // prefetch this step's emissions (consumed after WMMA -> latency overlap)
        float4 em0[3], em1[3];
        const float* ebt = eb + t * K_;
        #pragma unroll
        for (int m = 0; m < 3; ++m) {
            em0[m] = *(const float4*)(ebt + m * 16 + hf * 8);
            em1[m] = *(const float4*)(ebt + m * 16 + hf * 8 + 4);
        }
        // per-column max (one cross-lane swap)
        float mx = a[0][0];
        #pragma unroll
        for (int m = 0; m < 3; ++m)
            #pragma unroll
            for (int r = 0; r < 8; ++r) mx = fmaxf(mx, a[m][r]);
        mx = fmaxf(mx, __shfl_xor(mx, 16, 32));

        // v = exp(alpha - mx), packed to f16 pairs; swap halves across lanes (12 shuffles)
        unsigned p[3][4], q[3][4];
        #pragma unroll
        for (int m = 0; m < 3; ++m)
            #pragma unroll
            for (int u = 0; u < 4; ++u)
                p[m][u] = pk2(fast_exp(a[m][2 * u] - mx), fast_exp(a[m][2 * u + 1] - mx));
        #pragma unroll
        for (int m = 0; m < 3; ++m)
            #pragma unroll
            for (int u = 0; u < 4; ++u) q[m][u] = __shfl_xor(p[m][u], 16, 32);

        // assemble B fragments (32x16 f16, lanes<16: K=0..15 / lanes>=16: K=16..31)
        V16HU B0, B1;
        #pragma unroll
        for (int u = 0; u < 4; ++u) {
            B0.u[u]     = (hf == 0) ? p[0][u] : q[1][u];
            B0.u[u + 4] = (hf == 0) ? q[0][u] : p[1][u];
            B1.u[u]     = (hf == 0) ? p[2][u] : 0u;     // K>=48 pad rows are zero
            B1.u[u + 4] = (hf == 0) ? q[2][u] : 0u;
        }

        // s = E * v via 6 WMMAs; alpha_new = mx + maxT + ln(s) + emis
        #pragma unroll
        for (int m = 0; m < 3; ++m) {
            v8f c = {};
            c = __builtin_amdgcn_wmma_f32_16x16x32_f16(false, A[m][0], false, B0.v,
                                                       (short)0, c, false, false);
            c = __builtin_amdgcn_wmma_f32_16x16x32_f16(false, A[m][1], false, B1.v,
                                                       (short)0, c, false, false);
            const float base = mx + maxT;
            const float er[8] = { em0[m].x, em0[m].y, em0[m].z, em0[m].w,
                                  em1[m].x, em1[m].y, em1[m].z, em1[m].w };
            #pragma unroll
            for (int r = 0; r < 8; ++r)
                a[m][r] = base + fast_log(c[r]) + er[r];
        }
        // snapshot alpha at t == len-1 (per-lane cndmask, no divergence)
        #pragma unroll
        for (int m = 0; m < 3; ++m)
            #pragma unroll
            for (int r = 0; r < 8; ++r)
                af[m][r] = (t == t_last) ? a[m][r] : af[m][r];
    }

    // logZ[b] = logsumexp_i(af[i] + finalT[i])
    float z[3][8];
    #pragma unroll
    for (int m = 0; m < 3; ++m) {
        const float4 f0 = *(const float4*)(finalT + m * 16 + hf * 8);
        const float4 f1 = *(const float4*)(finalT + m * 16 + hf * 8 + 4);
        const float fr[8] = { f0.x, f0.y, f0.z, f0.w, f1.x, f1.y, f1.z, f1.w };
        #pragma unroll
        for (int r = 0; r < 8; ++r) z[m][r] = af[m][r] + fr[r];
    }
    float mz = z[0][0];
    #pragma unroll
    for (int m = 0; m < 3; ++m)
        #pragma unroll
        for (int r = 0; r < 8; ++r) mz = fmaxf(mz, z[m][r]);
    mz = fmaxf(mz, __shfl_xor(mz, 16, 32));
    float s = 0.0f;
    #pragma unroll
    for (int m = 0; m < 3; ++m)
        #pragma unroll
        for (int r = 0; r < 8; ++r) s += fast_exp(z[m][r] - mz);
    s += __shfl_xor(s, 16, 32);
    if (hf == 0) logZ[b] = mz + fast_log(s);
}

// ---------------- Kernel 3: gold-path score (one wave per batch element) ---------------
__global__ __launch_bounds__(32) void crf_path(
        const float* __restrict__ emis, const int* __restrict__ lengths,
        const int* __restrict__ tags, const float* __restrict__ prior,
        const float* __restrict__ trans, const float* __restrict__ finalT,
        float* __restrict__ path) {
    const int b = blockIdx.x, lane = threadIdx.x;
    int lenb = lengths[b]; if (lenb < 1) lenb = 1;
    const int* tb = tags + (size_t)b * T_;
    const float* eb = emis + (size_t)b * T_ * K_;
    float s = 0.0f;
    for (int t = lane; t < lenb; t += 32) {
        const int tg = tb[t];
        float sc = eb[(size_t)t * K_ + tg];
        sc += (t == 0) ? prior[tg] : trans[tg * K_ + tb[t - 1]];
        s += sc;
    }
    #pragma unroll
    for (int off = 16; off > 0; off >>= 1) s += __shfl_xor(s, off, 32);
    if (lane == 0) path[b] = s + finalT[tb[lenb - 1]];
}

// ---------------- Kernel 4: mean(logZ - path) ------------------------------------------
__global__ void crf_reduce(const float* __restrict__ logZ, const float* __restrict__ path,
                           float* __restrict__ out) {
    __shared__ float red[256];
    const int tid = threadIdx.x;
    float s = 0.0f;
    for (int b = tid; b < B_; b += 256) s += logZ[b] - path[b];
    red[tid] = s; __syncthreads();
    for (int k = 128; k > 0; k >>= 1) {
        if (tid < k) red[tid] += red[tid + k];
        __syncthreads();
    }
    if (tid == 0) out[0] = red[0] * (1.0f / (float)B_);
}

extern "C" void kernel_launch(void* const* d_in, const int* in_sizes, int n_in,
                              void* d_out, int out_size, void* d_ws, size_t ws_size,
                              hipStream_t stream) {
    const float* emis   = (const float*)d_in[0];
    const int*   lens   = (const int*)d_in[1];
    const int*   tags   = (const int*)d_in[2];
    const float* prior  = (const float*)d_in[3];
    const float* trans  = (const float*)d_in[4];
    const float* finalT = (const float*)d_in[5];

    char* ws = (char*)d_ws;
    float*          maxT  = (float*)(ws + 0);
    unsigned short* Ef16  = (unsigned short*)(ws + 256);          // 48*64*2 = 6144 B
    float*          logZ  = (float*)(ws + 8192);                  // 512 floats
    float*          path  = (float*)(ws + 8192 + 2048);           // 512 floats

    crf_prep<<<1, 256, 0, stream>>>(trans, maxT, Ef16);
    crf_forward<<<B_ / 16, 32, 0, stream>>>(emis, lens, Ef16, maxT, prior, finalT, logZ);
    crf_path<<<B_, 32, 0, stream>>>(emis, lens, tags, prior, trans, finalT, path);
    crf_reduce<<<1, 256, 0, stream>>>(logZ, path, (float*)d_out);
}
